// LightAttentionalGNN_88210038326004
// MI455X (gfx1250) — compile-verified, matrix operand order checked by reading, not run on hardware
//
#include <hip/hip_runtime.h>
#include <hip/hip_bf16.h>

// ---------------- problem constants ----------------
#define BB   2
#define DD   256
#define HH   4
#define HDD  64
#define LL   6
#define HIDD 64
#define NPTS 1536

typedef __attribute__((ext_vector_type(16))) _Float16 v16h;
typedef __attribute__((ext_vector_type(8)))  _Float16 v8h;
typedef __attribute__((ext_vector_type(4)))  _Float16 v4h;
typedef __attribute__((ext_vector_type(8)))  float    v8f;

#define LDST 40   // LDS row stride in halves (80 B, 16-B aligned)

// ---------------- tile stage: global fp32 -> LDS f16 ----------------
// Tile is 64 rows x 32 cols (row-major, stride LDST halves).
// FASTROW==0: source fast dim is the k (col) axis:  addr=(slow+r)*ld+fast+c
// FASTROW==1: source fast dim is the row axis:      addr=(slow+c)*ld+fast+r
template <int FASTROW>
__device__ __forceinline__ void load_tile(const float* __restrict__ src, int ld,
                                          int slowBase, int fastBase,
                                          _Float16* __restrict__ dst, int tid)
{
    float4 t[4];
    if (FASTROW == 0) {
#pragma unroll
        for (int e = 0; e < 4; ++e) {
            int lin = tid + e * 128;              // 0..511
            int r = lin >> 3, c4 = (lin & 7) << 2;
            t[e] = *(const float4*)(src + (size_t)(slowBase + r) * ld + fastBase + c4);
        }
#pragma unroll
        for (int e = 0; e < 4; ++e) {
            int lin = tid + e * 128;
            int r = lin >> 3, c4 = (lin & 7) << 2;
            v4h h = { (_Float16)t[e].x, (_Float16)t[e].y,
                      (_Float16)t[e].z, (_Float16)t[e].w };
            *(v4h*)(dst + (size_t)r * LDST + c4) = h;   // ds_store_b64
        }
    } else {
#pragma unroll
        for (int e = 0; e < 4; ++e) {
            int lin = tid + e * 128;              // 0..511
            int c = lin >> 4, r4 = (lin & 15) << 2;
            t[e] = *(const float4*)(src + (size_t)(slowBase + c) * ld + fastBase + r4);
        }
#pragma unroll
        for (int e = 0; e < 4; ++e) {
            int lin = tid + e * 128;
            int c = lin >> 4, r4 = (lin & 15) << 2;
            dst[(size_t)(r4 + 0) * LDST + c] = (_Float16)t[e].x;
            dst[(size_t)(r4 + 1) * LDST + c] = (_Float16)t[e].y;
            dst[(size_t)(r4 + 2) * LDST + c] = (_Float16)t[e].z;
            dst[(size_t)(r4 + 3) * LDST + c] = (_Float16)t[e].w;
        }
    }
}

// 16-half fragment = two aligned 16-B LDS reads (k koff..koff+7, 16+koff..23+koff)
__device__ __forceinline__ v16h frag(const _Float16* __restrict__ rowp, int koff)
{
    v8h lo = *(const v8h*)(rowp + koff);
    v8h hi = *(const v8h*)(rowp + 16 + koff);
    return __builtin_shufflevector(lo, hi, 0, 1, 2, 3, 4, 5, 6, 7,
                                   8, 9, 10, 11, 12, 13, 14, 15);
}

// ---------------- templated WMMA GEMM ----------------
// C[M,N] = (BE ? C : 0) + alpha*( opA(A) @ opB(B) + bias )
// TA: A stored (K,M) row-major. TB: B stored (N,K) row-major.
// BM: 0 none, 1 bias per-row(M), 2 bias per-col(N). M,N multiples of 64, K of 32.
template <int TA, int TB, int BM, bool BE>
__global__ __launch_bounds__(128)
void k_gemm(const float* __restrict__ A, int lda,
            const float* __restrict__ B, int ldb,
            float* __restrict__ C, int ldc,
            const float* __restrict__ bias,
            float alpha, int M, int N, int K)
{
    __shared__ __align__(16) _Float16 As[64 * LDST];
    __shared__ __align__(16) _Float16 Bs[64 * LDST];   // Bs[n][k]

    const int tid  = threadIdx.x;
    const int lane = tid & 31;
    const int wave = tid >> 5;
    const int wr = (wave >> 1) * 32;     // wave quadrant inside 64x64 block tile
    const int wc = (wave & 1) * 32;
    const int rowBase = blockIdx.y * 64;
    const int colBase = blockIdx.x * 64;

    const v8f vzero = {0.f, 0.f, 0.f, 0.f, 0.f, 0.f, 0.f, 0.f};
    v8f acc00 = vzero, acc01 = vzero, acc10 = vzero, acc11 = vzero;

    const int fr   = lane & 15;
    const int koff = (lane >> 4) << 3;   // 0 or 8

    for (int k0 = 0; k0 < K; k0 += 32) {
        if (TA == 0) load_tile<0>(A, lda, rowBase, k0, As, tid);
        else         load_tile<1>(A, lda, k0, rowBase, As, tid);
        if (TB == 1) load_tile<0>(B, ldb, colBase, k0, Bs, tid);
        else         load_tile<1>(B, ldb, k0, colBase, Bs, tid);
        __syncthreads();

        v16h a0 = frag(&As[(size_t)(wr + fr) * LDST], koff);
        v16h a1 = frag(&As[(size_t)(wr + 16 + fr) * LDST], koff);
        v16h b0 = frag(&Bs[(size_t)(wc + fr) * LDST], koff);
        v16h b1 = frag(&Bs[(size_t)(wc + 16 + fr) * LDST], koff);

        acc00 = __builtin_amdgcn_wmma_f32_16x16x32_f16(false, a0, false, b0, (short)0, acc00, false, false);
        acc01 = __builtin_amdgcn_wmma_f32_16x16x32_f16(false, a0, false, b1, (short)0, acc01, false, false);
        acc10 = __builtin_amdgcn_wmma_f32_16x16x32_f16(false, a1, false, b0, (short)0, acc10, false, false);
        acc11 = __builtin_amdgcn_wmma_f32_16x16x32_f16(false, a1, false, b1, (short)0, acc11, false, false);
        __syncthreads();
    }

    // C/D layout: element v -> M = v + (lane>=16 ? 8:0), N = lane&15
    const int nBase = colBase + wc + fr;
    const int mOff  = rowBase + wr + ((lane >> 4) << 3);
    const v8f* accs[4] = { &acc00, &acc01, &acc10, &acc11 };
#pragma unroll
    for (int ti = 0; ti < 2; ++ti) {
#pragma unroll
        for (int tj = 0; tj < 2; ++tj) {
            const v8f a = *accs[ti * 2 + tj];
            const int n  = nBase + tj * 16;
            const int mB = mOff + ti * 16;
#pragma unroll
            for (int v = 0; v < 8; ++v) {
                int m = mB + v;
                float r = a[v];
                if (BM == 1) r += bias[m];
                if (BM == 2) r += bias[n];
                r *= alpha;
                size_t idx = (size_t)m * ldc + n;
                if (BE) r += C[idx];
                C[idx] = r;
            }
        }
    }
}

template <int TA, int TB, int BM, bool BE>
static inline void gemm(hipStream_t s, const float* A, int lda, const float* B, int ldb,
                        float* C, int ldc, const float* bias, float alpha,
                        int M, int N, int K)
{
    dim3 g(N / 64, M / 64, 1);
    k_gemm<TA, TB, BM, BE><<<g, 128, 0, s>>>(A, lda, B, ldb, C, ldc, bias, alpha, M, N, K);
}

// ---------------- RoPE on (B, D, N) with channel c = d*H + h ----------------
__global__ __launch_bounds__(256)
void k_rope(float* __restrict__ t, const float* __restrict__ enc)
{
    int idx = blockIdx.x * blockDim.x + threadIdx.x;
    const int total = BB * HH * NPTS * (HDD / 2);
    if (idx >= total) return;
    int p = idx & 31;                       // HD/2 = 32 pairs
    int n = (idx >> 5) % NPTS;
    int h = (idx / (32 * NPTS)) % HH;
    int b = idx / (32 * NPTS * HH);
    int c0 = (2 * p) * HH + h;
    size_t base = (size_t)b * DD * NPTS + n;
    float x0 = t[base + (size_t)c0 * NPTS];
    float x1 = t[base + (size_t)(c0 + HH) * NPTS];
    float cs = enc[(size_t)n * HDD + 2 * p];
    float sn = enc[(size_t)NPTS * HDD + (size_t)n * HDD + 2 * p];
    t[base + (size_t)c0 * NPTS]        = x0 * cs - x1 * sn;
    t[base + (size_t)(c0 + HH) * NPTS] = x1 * cs + x0 * sn;
}

// ---------------- row softmax in place ----------------
__global__ __launch_bounds__(256)
void k_row_softmax(float* __restrict__ S, int cols)
{
    __shared__ float red[256];
    float* r = S + (size_t)blockIdx.x * cols;
    float mx = -3.4e38f;
    for (int j = threadIdx.x; j < cols; j += 256) mx = fmaxf(mx, r[j]);
    red[threadIdx.x] = mx; __syncthreads();
    for (int s = 128; s > 0; s >>= 1) {
        if (threadIdx.x < s) red[threadIdx.x] = fmaxf(red[threadIdx.x], red[threadIdx.x + s]);
        __syncthreads();
    }
    mx = red[0]; __syncthreads();
    float sum = 0.f;
    for (int j = threadIdx.x; j < cols; j += 256) {
        float e = __expf(r[j] - mx); r[j] = e; sum += e;
    }
    red[threadIdx.x] = sum; __syncthreads();
    for (int s = 128; s > 0; s >>= 1) {
        if (threadIdx.x < s) red[threadIdx.x] += red[threadIdx.x + s];
        __syncthreads();
    }
    float inv = 1.f / red[0];
    for (int j = threadIdx.x; j < cols; j += 256) r[j] *= inv;
}

// ---------------- BN (running stats) + ReLU ----------------
__global__ __launch_bounds__(256)
void k_bnrelu(float* __restrict__ h, const float* __restrict__ g,
              const float* __restrict__ bt)
{
    int idx = blockIdx.x * blockDim.x + threadIdx.x;
    const int total = BB * HIDD * NPTS;
    if (idx >= total) return;
    int c = (idx / NPTS) % HIDD;
    float x = h[idx];
    x = g[c] * x * rsqrtf(1.f + 1e-5f) + bt[c];
    h[idx] = fmaxf(x, 0.f);
}

// ---------------- matchability dot + sigmoid ----------------
__global__ __launch_bounds__(256)
void k_dot_sig(const float* __restrict__ dsc, const float* __restrict__ w,
               const float* __restrict__ bz, int bi,
               float* __restrict__ outSig, float* __restrict__ outRaw)
{
    int idx = blockIdx.x * blockDim.x + threadIdx.x;
    if (idx >= BB * NPTS) return;
    int b = idx / NPTS, n = idx % NPTS;
    const float* p = dsc + (size_t)b * DD * NPTS + n;
    float s = bz[bi];
#pragma unroll 4
    for (int c = 0; c < DD; ++c) s += w[c] * p[(size_t)c * NPTS];
    if (outRaw) outRaw[idx] = s;
    outSig[idx] = 1.f / (1.f + __expf(-s));
}

// ---------------- sim init: nn_sim * nn_weight ----------------
__global__ __launch_bounds__(256)
void k_siminit(const float* __restrict__ nn_sim, const float* __restrict__ wsc,
               float* __restrict__ out, size_t total)
{
    size_t idx = (size_t)blockIdx.x * blockDim.x + threadIdx.x;
    if (idx >= total) return;
    out[idx] = nn_sim[idx] * wsc[0];
}

// ---------------- log-softmax stats ----------------
__global__ __launch_bounds__(256)
void k_row_stats(const float* __restrict__ sim, float* __restrict__ rmax,
                 float* __restrict__ rsum, int cols)
{
    __shared__ float red[256];
    const float* r = sim + (size_t)blockIdx.x * cols;
    float mx = -3.4e38f;
    for (int j = threadIdx.x; j < cols; j += 256) mx = fmaxf(mx, r[j]);
    red[threadIdx.x] = mx; __syncthreads();
    for (int s = 128; s > 0; s >>= 1) {
        if (threadIdx.x < s) red[threadIdx.x] = fmaxf(red[threadIdx.x], red[threadIdx.x + s]);
        __syncthreads();
    }
    mx = red[0]; __syncthreads();
    float sum = 0.f;
    for (int j = threadIdx.x; j < cols; j += 256) sum += __expf(r[j] - mx);
    red[threadIdx.x] = sum; __syncthreads();
    for (int s = 128; s > 0; s >>= 1) {
        if (threadIdx.x < s) red[threadIdx.x] += red[threadIdx.x + s];
        __syncthreads();
    }
    if (threadIdx.x == 0) { rmax[blockIdx.x] = mx; rsum[blockIdx.x] = red[0]; }
}

__global__ __launch_bounds__(256)
void k_col_stats(const float* __restrict__ sim, float* __restrict__ cmax,
                 float* __restrict__ csum, int rows, int cols)
{
    __shared__ float red[256];
    int b = blockIdx.x / cols, j = blockIdx.x % cols;
    const float* base = sim + (size_t)b * rows * cols + j;
    float mx = -3.4e38f;
    for (int i = threadIdx.x; i < rows; i += 256) mx = fmaxf(mx, base[(size_t)i * cols]);
    red[threadIdx.x] = mx; __syncthreads();
    for (int s = 128; s > 0; s >>= 1) {
        if (threadIdx.x < s) red[threadIdx.x] = fmaxf(red[threadIdx.x], red[threadIdx.x + s]);
        __syncthreads();
    }
    mx = red[0]; __syncthreads();
    float sum = 0.f;
    for (int i = threadIdx.x; i < rows; i += 256) sum += __expf(base[(size_t)i * cols] - mx);
    red[threadIdx.x] = sum; __syncthreads();
    for (int s = 128; s > 0; s >>= 1) {
        if (threadIdx.x < s) red[threadIdx.x] += red[threadIdx.x + s];
        __syncthreads();
    }
    if (threadIdx.x == 0) { cmax[blockIdx.x] = mx; csum[blockIdx.x] = red[0]; }
}

__device__ __forceinline__ float logsigf(float x)
{
    return fminf(x, 0.f) - log1pf(__expf(-fabsf(x)));
}

// ---------------- final (M+1)x(N+1) score assembly ----------------
__global__ __launch_bounds__(256)
void k_assemble(const float* __restrict__ sim,
                const float* __restrict__ rmax, const float* __restrict__ rsum,
                const float* __restrict__ cmax, const float* __restrict__ csum,
                const float* __restrict__ z0, const float* __restrict__ z1,
                float* __restrict__ scores)
{
    const int MP = NPTS + 1;
    size_t idx = (size_t)blockIdx.x * blockDim.x + threadIdx.x;
    size_t total = (size_t)BB * MP * MP;
    if (idx >= total) return;
    int j = idx % MP;
    int i = (idx / MP) % MP;
    int b = (int)(idx / ((size_t)MP * MP));
    float val;
    if (i < NPTS && j < NPTS) {
        float s  = sim[(size_t)b * NPTS * NPTS + (size_t)i * NPTS + j];
        float s0 = s - rmax[b * NPTS + i] - __logf(rsum[b * NPTS + i]);
        float s1 = s - cmax[b * NPTS + j] - __logf(csum[b * NPTS + j]);
        val = s0 + s1 + logsigf(z0[b * NPTS + i]) + logsigf(z1[b * NPTS + j]);
    } else if (i < NPTS) {
        val = logsigf(-z0[b * NPTS + i]);
    } else if (j < NPTS) {
        val = logsigf(-z1[b * NPTS + j]);
    } else {
        val = 0.f;
    }
    scores[idx] = val;
}

// =======================================================================
extern "C" void kernel_launch(void* const* d_in, const int* in_sizes, int n_in,
                              void* d_out, int out_size, void* d_ws, size_t ws_size,
                              hipStream_t stream)
{
    (void)in_sizes; (void)n_in; (void)out_size; (void)ws_size;

    const float* desc0   = (const float*)d_in[0];
    const float* desc1   = (const float*)d_in[1];
    const float* enc0    = (const float*)d_in[2];
    const float* enc1    = (const float*)d_in[3];
    const float* nn_sim  = (const float*)d_in[4];
    const float* Wq = (const float*)d_in[5];  const float* bq = (const float*)d_in[6];
    const float* Wk = (const float*)d_in[7];  const float* bk = (const float*)d_in[8];
    const float* Wv = (const float*)d_in[9];  const float* bv = (const float*)d_in[10];
    const float* Wm = (const float*)d_in[11]; const float* bm = (const float*)d_in[12];
    const float* W1 = (const float*)d_in[13]; const float* b1 = (const float*)d_in[14];
    const float* gm = (const float*)d_in[15]; const float* bt = (const float*)d_in[16];
    const float* W2 = (const float*)d_in[17]; const float* b2 = (const float*)d_in[18];
    const float* Wz = (const float*)d_in[19]; const float* bz = (const float*)d_in[20];
    const float* Wfp = (const float*)d_in[21]; const float* bfp = (const float*)d_in[22];
    const float* nn_w = (const float*)d_in[23];

    // ----- output layout (flat, return order) -----
    float* out = (float*)d_out;
    const size_t nScores = (size_t)BB * (NPTS + 1) * (NPTS + 1);
    float* out_scores = out;
    float* out_sim    = out_scores + nScores;
    float* out_z0     = out_sim + (size_t)BB * NPTS * NPTS;
    float* out_z1     = out_z0 + BB * NPTS;
    float* out_i0     = out_z1 + BB * NPTS;
    float* out_i1     = out_i0 + 2 * BB * NPTS;

    // ----- workspace carve-out -----
    float* ws = (float*)d_ws;
    size_t o = 0;
    auto alloc = [&](size_t n) { float* p = ws + o; o += n; return p; };
    const size_t DN = (size_t)BB * DD * NPTS;
    float* dw0  = alloc(DN);
    float* dw1  = alloc(DN);
    float* qb   = alloc(DN);
    float* kb   = alloc(DN);
    float* vb   = alloc(DN);
    float* attn = alloc(DN);
    float* msg0 = alloc(DN);
    float* msg1 = alloc(DN);
    float* hb   = alloc((size_t)BB * HIDD * NPTS);
    float* prob = alloc((size_t)NPTS * NPTS);
    float* md0  = alloc((size_t)BB * NPTS * DD);
    float* md1  = alloc((size_t)BB * NPTS * DD);
    float* z0r  = alloc(BB * NPTS);
    float* z1r  = alloc(BB * NPTS);
    float* rmx  = alloc(BB * NPTS);
    float* rsm  = alloc(BB * NPTS);
    float* cmx  = alloc(BB * NPTS);
    float* csm  = alloc(BB * NPTS);

    // copy descriptors into mutable workspace
    hipMemcpyAsync(dw0, desc0, DN * sizeof(float), hipMemcpyDeviceToDevice, stream);
    hipMemcpyAsync(dw1, desc1, DN * sizeof(float), hipMemcpyDeviceToDevice, stream);

    const int dnb = DD * NPTS;   // per-batch stride

    for (int i = 0; i < LL; ++i) {
        const bool cross = (i & 1);
        const float* Wq_i = Wq + (size_t)i * DD * DD; const float* bq_i = bq + (size_t)i * DD;
        const float* Wk_i = Wk + (size_t)i * DD * DD; const float* bk_i = bk + (size_t)i * DD;
        const float* Wv_i = Wv + (size_t)i * DD * DD; const float* bv_i = bv + (size_t)i * DD;
        const float* Wm_i = Wm + (size_t)i * DD * DD; const float* bm_i = bm + (size_t)i * DD;
        const float* W1_i = W1 + (size_t)i * HIDD * 2 * DD; const float* b1_i = b1 + (size_t)i * HIDD;
        const float* g_i  = gm + (size_t)i * HIDD; const float* bt_i = bt + (size_t)i * HIDD;
        const float* W2_i = W2 + (size_t)i * DD * HIDD; const float* b2_i = b2 + (size_t)i * DD;

        // ---- messages for both descriptor sets (before any update) ----
        for (int t = 0; t < 2; ++t) {
            float* x   = t ? dw1 : dw0;
            float* src = cross ? (t ? dw0 : dw1) : x;
            const float* enc = t ? enc1 : enc0;
            float* msg = t ? msg1 : msg0;

            for (int b = 0; b < BB; ++b) {
                gemm<0,0,1,false>(stream, Wq_i, DD, x   + b * dnb, NPTS,
                                  qb + b * dnb, NPTS, bq_i, 1.f, DD, NPTS, DD);
                gemm<0,0,1,false>(stream, Wk_i, DD, src + b * dnb, NPTS,
                                  kb + b * dnb, NPTS, bk_i, 1.f, DD, NPTS, DD);
                gemm<0,0,1,false>(stream, Wv_i, DD, src + b * dnb, NPTS,
                                  vb + b * dnb, NPTS, bv_i, 1.f, DD, NPTS, DD);
            }
            if (!cross) {
                int tot = BB * HH * NPTS * (HDD / 2);
                k_rope<<<(tot + 255) / 256, 256, 0, stream>>>(qb, enc);
                k_rope<<<(tot + 255) / 256, 256, 0, stream>>>(kb, enc);
            }
            for (int b = 0; b < BB; ++b) {
                for (int h = 0; h < HH; ++h) {
                    const float* Qh = qb + (size_t)b * dnb + (size_t)h * NPTS;  // (64,NPTS), lda=H*NPTS
                    const float* Kh = kb + (size_t)b * dnb + (size_t)h * NPTS;
                    const float* Vh = vb + (size_t)b * dnb + (size_t)h * NPTS;
                    // scores (n,m) = Q^T K / sqrt(HD)
                    gemm<1,0,0,false>(stream, Qh, HH * NPTS, Kh, HH * NPTS,
                                      prob, NPTS, nullptr, 0.125f, NPTS, NPTS, HDD);
                    k_row_softmax<<<NPTS, 256, 0, stream>>>(prob, NPTS);
                    // out (hd,n) = V @ P^T
                    gemm<0,1,0,false>(stream, Vh, HH * NPTS, prob, NPTS,
                                      attn + (size_t)b * dnb + (size_t)h * NPTS, HH * NPTS,
                                      nullptr, 1.f, HDD, NPTS, NPTS);
                }
            }
            for (int b = 0; b < BB; ++b)
                gemm<0,0,1,false>(stream, Wm_i, DD, attn + b * dnb, NPTS,
                                  msg + b * dnb, NPTS, bm_i, 1.f, DD, NPTS, DD);
        }

        // ---- MLP on concat([d, msg]) + residual, for both sets ----
        for (int t = 0; t < 2; ++t) {
            float* dt  = t ? dw1 : dw0;
            float* msg = t ? msg1 : msg0;
            for (int b = 0; b < BB; ++b) {
                gemm<0,0,1,false>(stream, W1_i,      2 * DD, dt  + b * dnb, NPTS,
                                  hb + (size_t)b * HIDD * NPTS, NPTS, b1_i, 1.f, HIDD, NPTS, DD);
                gemm<0,0,0,true >(stream, W1_i + DD, 2 * DD, msg + b * dnb, NPTS,
                                  hb + (size_t)b * HIDD * NPTS, NPTS, nullptr, 1.f, HIDD, NPTS, DD);
            }
            int tot = BB * HIDD * NPTS;
            k_bnrelu<<<(tot + 255) / 256, 256, 0, stream>>>(hb, g_i, bt_i);
            for (int b = 0; b < BB; ++b)
                gemm<0,0,1,true>(stream, W2_i, HIDD, hb + (size_t)b * HIDD * NPTS, NPTS,
                                 dt + b * dnb, NPTS, b2_i, 1.f, DD, NPTS, HIDD);
        }

        // ---- width-confidence intermediates ----
        if (cross && i < LL - 1) {
            int idx = i / 2;  // i=1 -> 0, i=3 -> 1
            int blks = (BB * NPTS + 255) / 256;
            k_dot_sig<<<blks, 256, 0, stream>>>(dw0, Wz + (size_t)i * DD, bz, i,
                                                out_i0 + (size_t)idx * BB * NPTS, nullptr);
            k_dot_sig<<<blks, 256, 0, stream>>>(dw1, Wz + (size_t)i * DD, bz, i,
                                                out_i1 + (size_t)idx * BB * NPTS, nullptr);
        }
    }

    // ---- final projection: md = (d^T @ Wfp^T + bfp) / D^0.25 (D^0.25 = 4) ----
    for (int b = 0; b < BB; ++b) {
        gemm<1,1,2,false>(stream, dw0 + b * dnb, NPTS, Wfp, DD,
                          md0 + (size_t)b * NPTS * DD, DD, bfp, 0.25f, NPTS, DD, DD);
        gemm<1,1,2,false>(stream, dw1 + b * dnb, NPTS, Wfp, DD,
                          md1 + (size_t)b * NPTS * DD, DD, bfp, 0.25f, NPTS, DD, DD);
    }

    // ---- sim = md0 @ md1^T + nn_sim * nn_weight (written into output) ----
    {
        size_t tot = (size_t)BB * NPTS * NPTS;
        k_siminit<<<(int)((tot + 255) / 256), 256, 0, stream>>>(nn_sim, nn_w, out_sim, tot);
        for (int b = 0; b < BB; ++b)
            gemm<0,1,0,true>(stream, md0 + (size_t)b * NPTS * DD, DD,
                             md1 + (size_t)b * NPTS * DD, DD,
                             out_sim + (size_t)b * NPTS * NPTS, NPTS,
                             nullptr, 1.f, NPTS, NPTS, DD);
    }

    // ---- matchability logits ----
    {
        int blks = (BB * NPTS + 255) / 256;
        k_dot_sig<<<blks, 256, 0, stream>>>(dw0, Wz + (size_t)(LL - 1) * DD, bz, LL - 1, out_z0, z0r);
        k_dot_sig<<<blks, 256, 0, stream>>>(dw1, Wz + (size_t)(LL - 1) * DD, bz, LL - 1, out_z1, z1r);
    }

    // ---- log-softmax stats + score assembly ----
    k_row_stats<<<BB * NPTS, 256, 0, stream>>>(out_sim, rmx, rsm, NPTS);
    k_col_stats<<<BB * NPTS, 256, 0, stream>>>(out_sim, cmx, csm, NPTS, NPTS);
    {
        size_t tot = (size_t)BB * (NPTS + 1) * (NPTS + 1);
        k_assemble<<<(int)((tot + 255) / 256), 256, 0, stream>>>(
            out_sim, rmx, rsm, cmx, csm, z0r, z1r, out_scores);
    }
}
